// HBitLinear_90967407329362
// MI455X (gfx1250) — 8-bit matrix operand order repaired and checked by reading, compiled, not run on hardware
//
#include <hip/hip_runtime.h>


typedef int v8i __attribute__((ext_vector_type(8)));

#define D_IN   2048
#define D_OUT  8192
#define NROWS  8192
#define LN_EPS 1e-5f

#define WS_PARTIALS 0
#define WS_WSCALE   16384
#define WS_WQ       32768
#define WS_AHI      (WS_WQ  + D_OUT * D_IN)
#define WS_ALO      (WS_AHI + NROWS * D_IN)

__device__ __forceinline__ float warpSum(float v) {
  #pragma unroll
  for (int o = 16; o > 0; o >>= 1) v += __shfl_down(v, o, 32);
  return v;
}
__device__ __forceinline__ float warpMax(float v) {
  #pragma unroll
  for (int o = 16; o > 0; o >>= 1) v = fmaxf(v, __shfl_down(v, o, 32));
  return v;
}

__global__ __launch_bounds__(256) void hbl_absw_partials(const float* __restrict__ w,
                                                         float* __restrict__ partials) {
  __shared__ float red[8];
  const long base = (long)blockIdx.x * 4096;
  float ps = 0.f;
  for (int j = threadIdx.x; j < 4096; j += 256) ps += fabsf(w[base + j]);
  ps = warpSum(ps);
  if ((threadIdx.x & 31) == 0) red[threadIdx.x >> 5] = ps;
  __syncthreads();
  if (threadIdx.x == 0) {
    float a = 0.f;
    #pragma unroll
    for (int i = 0; i < 8; ++i) a += red[i];
    partials[blockIdx.x] = a;
  }
}

__global__ __launch_bounds__(256) void hbl_scale_finalize(const float* __restrict__ partials,
                                                          float* __restrict__ wscale) {
  __shared__ float red[8];
  float ps = 0.f;
  for (int j = threadIdx.x; j < 4096; j += 256) ps += partials[j];
  ps = warpSum(ps);
  if ((threadIdx.x & 31) == 0) red[threadIdx.x >> 5] = ps;
  __syncthreads();
  if (threadIdx.x == 0) {
    float a = 0.f;
    #pragma unroll
    for (int i = 0; i < 8; ++i) a += red[i];
    wscale[0] = a * (1.0f / 16777216.0f);
  }
}

__global__ __launch_bounds__(256) void hbl_ternarize(const float* __restrict__ w,
                                                     const float* __restrict__ wscale,
                                                     signed char* __restrict__ Wq) {
  const float th = 0.5f * wscale[0];
  const int n4 = (D_OUT * D_IN) / 4;
  for (int i4 = blockIdx.x * 256 + threadIdx.x; i4 < n4; i4 += 4096 * 256) {
    float4 v = ((const float4*)w)[i4];
    char4 q;
    q.x = v.x > th ? 1 : (v.x < -th ? -1 : 0);
    q.y = v.y > th ? 1 : (v.y < -th ? -1 : 0);
    q.z = v.z > th ? 1 : (v.z < -th ? -1 : 0);
    q.w = v.w > th ? 1 : (v.w < -th ? -1 : 0);
    ((char4*)Wq)[i4] = q;
  }
}

__global__ __launch_bounds__(256) void hbl_act(const float* __restrict__ x,
                                               const float* __restrict__ gamma,
                                               const float* __restrict__ beta,
                                               signed char* __restrict__ Ahi,
                                               signed char* __restrict__ Alo) {
  __shared__ float s[D_IN];
  __shared__ float red[8];
  const int t = threadIdx.x;
  const int lane = t & 31, wv = t >> 5;
  const long base = (long)blockIdx.x * D_IN;

  float ps = 0.f;
  for (int j = t; j < D_IN; j += 256) { float v = x[base + j]; s[j] = v; ps += v; }
  ps = warpSum(ps);
  if (lane == 0) red[wv] = ps;
  __syncthreads();
  if (t == 0) { float a = 0.f; for (int i = 0; i < 8; ++i) a += red[i]; red[0] = a; }
  __syncthreads();
  const float mu = red[0] * (1.0f / D_IN);
  __syncthreads();

  float pv = 0.f;
  for (int j = t; j < D_IN; j += 256) { float d = s[j] - mu; pv += d * d; }
  pv = warpSum(pv);
  if (lane == 0) red[wv] = pv;
  __syncthreads();
  if (t == 0) { float a = 0.f; for (int i = 0; i < 8; ++i) a += red[i]; red[0] = a; }
  __syncthreads();
  const float rs = rsqrtf(red[0] * (1.0f / D_IN) + LN_EPS);
  __syncthreads();

  float pm = 0.f;
  for (int j = t; j < D_IN; j += 256) {
    float xl = (s[j] - mu) * rs * gamma[j] + beta[j];
    s[j] = xl;
    pm = fmaxf(pm, fabsf(xl));
  }
  pm = warpMax(pm);
  if (lane == 0) red[wv] = pm;
  __syncthreads();
  if (t == 0) { float a = 0.f; for (int i = 0; i < 8; ++i) a = fmaxf(a, red[i]); red[0] = a; }
  __syncthreads();
  const float qs = 7.0f / fmaxf(red[0], 1e-5f);
  __syncthreads();

  for (int j = t; j < D_IN; j += 256) {
    float q = rintf(s[j] * qs);
    s[j] = fminf(fmaxf(q, -7.0f), 7.0f);
  }
  __syncthreads();

  for (int h = 1; h < D_IN; h <<= 1) {
    for (int p = t; p < D_IN / 2; p += 256) {
      int i = ((p & ~(h - 1)) << 1) | (p & (h - 1));
      float a = s[i], b = s[i + h];
      s[i] = a + b;
      s[i + h] = a - b;
    }
    __syncthreads();
  }

  for (int j = t; j < D_IN; j += 256) {
    int xh = (int)s[j];
    int h1 = (xh + 64) >> 7;
    int h0 = xh - (h1 << 7);
    Ahi[base + j] = (signed char)h1;
    Alo[base + j] = (signed char)h0;
  }
}

#define BM 64
#define BN 256
#define BK 64
#define LDA 80
#define LDB 80

__device__ __forceinline__ void async_b128(unsigned lds_addr, unsigned voff,
                                           const signed char* sbase) {
  asm volatile("global_load_async_to_lds_b128 %0, %1, %2"
               :: "v"(lds_addr), "v"(voff), "s"(sbase)
               : "memory");
}
__device__ __forceinline__ void wait_async0() {
  asm volatile("s_wait_asynccnt 0x0" ::: "memory");
}

__global__ __launch_bounds__(256) void hbl_gemm(const signed char* __restrict__ Ahi,
                                                const signed char* __restrict__ Alo,
                                                const signed char* __restrict__ Bq,
                                                float* __restrict__ out) {
  __shared__ __attribute__((aligned(16))) signed char sAhi[2][BM * LDA];
  __shared__ __attribute__((aligned(16))) signed char sAlo[2][BM * LDA];
  __shared__ __attribute__((aligned(16))) signed char sB[2][BN * LDB];

  const int t = threadIdx.x;
  const int lane = t & 31;
  const int wave = t >> 5;
  const int wm = wave >> 2;
  const int wn = wave & 3;
  const int laneLo = lane & 15;
  const int hb = lane >> 4;

  const int m0 = blockIdx.y * BM;
  const int n0 = blockIdx.x * BN;

  const int arow = t >> 2;
  const int acol = (t & 3) << 4;
  const unsigned aoff = (unsigned)((m0 + arow) * D_IN + acol);
  unsigned boff[4];
  unsigned ldsB_off[4];
  #pragma unroll
  for (int i = 0; i < 4; ++i) {
    const int c = t + (i << 8);
    boff[i] = (unsigned)((n0 + (c >> 2)) * D_IN + ((c & 3) << 4));
    ldsB_off[i] = (unsigned)((c >> 2) * LDB + ((c & 3) << 4));
  }
  const unsigned ldsAh = (unsigned)(unsigned long long)&sAhi[0][arow * LDA + acol];
  const unsigned ldsAl = (unsigned)(unsigned long long)&sAlo[0][arow * LDA + acol];
  const unsigned ldsB0 = (unsigned)(unsigned long long)&sB[0][0];

  v8i acc_hi[2][4], acc_lo[2][4];
  #pragma unroll
  for (int mi = 0; mi < 2; ++mi)
    #pragma unroll
    for (int ni = 0; ni < 4; ++ni) {
      acc_hi[mi][ni] = (v8i){0, 0, 0, 0, 0, 0, 0, 0};
      acc_lo[mi][ni] = (v8i){0, 0, 0, 0, 0, 0, 0, 0};
    }

  async_b128(ldsAh, aoff, Ahi);
  async_b128(ldsAl, aoff, Alo);
  #pragma unroll
  for (int i = 0; i < 4; ++i) async_b128(ldsB0 + ldsB_off[i], boff[i], Bq);
  wait_async0();
  __syncthreads();

  const int KSTEPS = D_IN / BK;
  for (int ks = 0; ks < KSTEPS; ++ks) {
    const int cur = ks & 1;
    const int nxt = cur ^ 1;

    if (ks + 1 < KSTEPS) {
      const int k0 = (ks + 1) * BK;
      async_b128(ldsAh + nxt * (BM * LDA), aoff, Ahi + k0);
      async_b128(ldsAl + nxt * (BM * LDA), aoff, Alo + k0);
      #pragma unroll
      for (int i = 0; i < 4; ++i)
        async_b128(ldsB0 + nxt * (BN * LDB) + ldsB_off[i], boff[i], Bq + k0);
    }

    v8i afh[2], afl[2];
    #pragma unroll
    for (int mi = 0; mi < 2; ++mi) {
      const int row = wm * 32 + mi * 16 + laneLo;
      const signed char* ph = &sAhi[cur][row * LDA + (hb << 4)];
      const signed char* pl = &sAlo[cur][row * LDA + (hb << 4)];
      #pragma unroll
      for (int g = 0; g < 4; ++g) {
        int2 dh = *(const int2*)(ph + (g & 1) * 8 + (g >> 1) * 32);
        int2 dl = *(const int2*)(pl + (g & 1) * 8 + (g >> 1) * 32);
        afh[mi][2 * g] = dh.x; afh[mi][2 * g + 1] = dh.y;
        afl[mi][2 * g] = dl.x; afl[mi][2 * g + 1] = dl.y;
      }
    }
    v8i bf[4];
    #pragma unroll
    for (int ni = 0; ni < 4; ++ni) {
      const int col = wn * 64 + ni * 16 + laneLo;
      const signed char* pbp = &sB[cur][col * LDB + (hb << 4)];
      int4 b0 = *(const int4*)(pbp);
      int4 b1 = *(const int4*)(pbp + 32);
      bf[ni][0] = b0.x; bf[ni][1] = b0.y; bf[ni][2] = b0.z; bf[ni][3] = b0.w;
      bf[ni][4] = b1.x; bf[ni][5] = b1.y; bf[ni][6] = b1.z; bf[ni][7] = b1.w;
    }
    #pragma unroll
    for (int ni = 0; ni < 4; ++ni)
      #pragma unroll
      for (int mi = 0; mi < 2; ++mi) {
        acc_hi[mi][ni] = __builtin_amdgcn_wmma_i32_16x16x64_iu8(
            true, afh[mi], true, bf[ni], acc_hi[mi][ni], false, false);
        acc_lo[mi][ni] = __builtin_amdgcn_wmma_i32_16x16x64_iu8(
            true, afl[mi], true, bf[ni], acc_lo[mi][ni], false, false);
      }

    wait_async0();
    __syncthreads();
  }

  #pragma unroll
  for (int mi = 0; mi < 2; ++mi) {
    const int rowBase = m0 + wm * 32 + mi * 16 + (hb << 3);
    #pragma unroll
    for (int ni = 0; ni < 4; ++ni) {
      const int col = n0 + wn * 64 + ni * 16 + laneLo;
      #pragma unroll
      for (int v = 0; v < 8; ++v) {
        int g = acc_hi[mi][ni][v] * 128 + acc_lo[mi][ni][v];
        out[(long)(rowBase + v) * D_OUT + col] = (float)g;
      }
    }
  }
}

__global__ __launch_bounds__(256) void hbl_fwht_out(float* __restrict__ out,
                                                    const float* __restrict__ wscale) {
  __shared__ float s[D_OUT];
  const long base = (long)blockIdx.x * D_OUT;
  for (int j = threadIdx.x; j < D_OUT; j += 256) s[j] = out[base + j];
  __syncthreads();
  for (int h = 1; h < D_OUT; h <<= 1) {
    for (int p = threadIdx.x; p < D_OUT / 2; p += 256) {
      int i = ((p & ~(h - 1)) << 1) | (p & (h - 1));
      float a = s[i], b = s[i + h];
      s[i] = a + b;
      s[i + h] = a - b;
    }
    __syncthreads();
  }
  const float sc = wscale[0];
  for (int j = threadIdx.x; j < D_OUT; j += 256) out[base + j] = s[j] * sc;
}

extern "C" void kernel_launch(void* const* d_in, const int* in_sizes, int n_in,
                              void* d_out, int out_size, void* d_ws, size_t ws_size,
                              hipStream_t stream) {
  (void)in_sizes; (void)n_in; (void)out_size; (void)ws_size;

  const float* x     = (const float*)d_in[0];
  const float* w     = (const float*)d_in[1];
  const float* gamma = (const float*)d_in[2];
  const float* beta  = (const float*)d_in[3];
  float* out = (float*)d_out;
  char*  ws  = (char*)d_ws;

  float*       partials = (float*)(ws + WS_PARTIALS);
  float*       wscale   = (float*)(ws + WS_WSCALE);
  signed char* Wq       = (signed char*)(ws + WS_WQ);
  signed char* Ahi      = (signed char*)(ws + WS_AHI);
  signed char* Alo      = (signed char*)(ws + WS_ALO);

  hbl_absw_partials<<<dim3(4096), dim3(256), 0, stream>>>(w, partials);
  hbl_scale_finalize<<<dim3(1), dim3(256), 0, stream>>>(partials, wscale);
  hbl_ternarize<<<dim3(4096), dim3(256), 0, stream>>>(w, wscale, Wq);
  hbl_act<<<dim3(NROWS), dim3(256), 0, stream>>>(x, gamma, beta, Ahi, Alo);
  hbl_gemm<<<dim3(D_OUT / BN, NROWS / BM), dim3(256), 0, stream>>>(Ahi, Alo, Wq, out);
  hbl_fwht_out<<<dim3(NROWS), dim3(256), 0, stream>>>(out, wscale);
}
